// TemporalWindowAttention_3186865733841
// MI455X (gfx1250) — matrix operand order repaired and checked by reading, not compiled, not run on hardware
//
#include <hip/hip_runtime.h>

// TemporalWindowAttention for MI455X (gfx1250, wave32, WMMA).
// Pipeline: cvt weights -> gather+LN -> QKV gemm (wmma f16, V stored
// transposed) -> attention (wmma f16, softmax f32 in LDS) -> proj gemm +
// residual + scatter.  Workspace ~113 MB (fits MI455X 192MB L2).

typedef __attribute__((ext_vector_type(16))) _Float16 v16h;
typedef __attribute__((ext_vector_type(8)))  float    v8f;

#define WINDOWS 128
#define L_TOK   256
#define CH      256
#define NH      8
#define HD      32
#define T_      4
#define IMG_HW  64
#define SCALE_F 0.17677669529663687f   // 1/sqrt(32)
#define EPS_F   1e-5f

// ---------------- WMMA fragment helpers (contiguous k: lane half h holds K = 16h..16h+15) ----------------

// A fragment: 16x32 f16 tile from row-major src (leading dim ld).
// lane<16: row M=lane, K = 0..15; lane>=16: row M=lane-16, K = 16..31
__device__ __forceinline__ v16h load_a(const _Float16* base, int ld, int row0, int k0) {
  int lane = threadIdx.x & 31;
  int hi = (lane >> 4) & 1;
  const _Float16* p = base + (size_t)(row0 + (lane & 15)) * ld + k0 + hi * 16;
  v16h a;
#pragma unroll
  for (int i = 0; i < 16; ++i) a[i] = p[i];
  return a;
}

// B fragment: 32x16 f16 tile where source is stored [N][K] row-major
// (column n of B is row n of the source) -> 16 contiguous halves per lane.
// lane<16: col N=lane, K=0..15; lane>=16: col N=lane-16, K=16..31
__device__ __forceinline__ v16h load_b_nk(const _Float16* base, int ld, int n0, int k0) {
  int lane = threadIdx.x & 31;
  int hi = (lane >> 4) & 1;
  const _Float16* p = base + (size_t)(n0 + (lane & 15)) * ld + k0 + hi * 16;
  v16h b;
#pragma unroll
  for (int i = 0; i < 16; ++i) b[i] = p[i];
  return b;
}

__device__ __forceinline__ v8f wmma_f32(v16h a, v16h b, v8f c) {
  return __builtin_amdgcn_wmma_f32_16x16x32_f16(false, a, false, b, (short)0, c,
                                                false, false);
}

// ---------------- K0: fp32 -> f16 weight conversion ----------------
__global__ void k_cvt_w(const float* __restrict__ wq, const float* __restrict__ wp,
                        _Float16* __restrict__ wq16, _Float16* __restrict__ wp16) {
  int i = blockIdx.x * blockDim.x + threadIdx.x;
  if (i < 768 * 256) {
    wq16[i] = (_Float16)wq[i];
  } else {
    int j = i - 768 * 256;
    if (j < 256 * 256) wp16[j] = (_Float16)wp[j];
  }
}

// ---------------- K1: window gather + LayerNorm ----------------
// One wave per token row (n,l): gathers 256 channels, shuffle-reduces
// mean/var, writes xw (fp32 residual copy) and xn (f16 normalized).
__global__ void k_gather_ln(const float* __restrict__ x, const float* __restrict__ gamma,
                            const float* __restrict__ beta, float* __restrict__ xw,
                            _Float16* __restrict__ xn) {
  int wave = threadIdx.x >> 5;
  int lane = threadIdx.x & 31;
  int row = blockIdx.x * 8 + wave;       // n*256 + l
  int n = row >> 8;
  int l = row & 255;
  int b = n >> 6, nh = (n >> 3) & 7, nw = n & 7;
  int t = l >> 6, wy = (l >> 3) & 7, wx = l & 7;
  size_t sp = (size_t)(nh * 8 + wy) * IMG_HW + (nw * 8 + wx);
  const float* xb = x + (size_t)(b * T_ + t) * CH * (IMG_HW * IMG_HW) + sp;

  float v[8];
  float s = 0.f, s2 = 0.f;
#pragma unroll
  for (int j = 0; j < 8; ++j) {
    int c = j * 32 + lane;
    float f = xb[(size_t)c * (IMG_HW * IMG_HW)];
    v[j] = f; s += f; s2 += f * f;
  }
#pragma unroll
  for (int off = 16; off > 0; off >>= 1) {
    s  += __shfl_xor(s,  off, 32);
    s2 += __shfl_xor(s2, off, 32);
  }
  float mean = s * (1.f / 256.f);
  float var  = s2 * (1.f / 256.f) - mean * mean;
  float rstd = rsqrtf(var + EPS_F);

  float*    xwr = xw + (size_t)row * CH;
  _Float16* xnr = xn + (size_t)row * CH;
#pragma unroll
  for (int j = 0; j < 8; ++j) {
    int c = j * 32 + lane;
    xwr[c] = v[j];
    xnr[c] = (_Float16)((v[j] - mean) * rstd * gamma[c] + beta[c]);
  }
}

// ---------------- K2: QKV GEMM (M=256,N=768,K=256) ----------------
// One wave per 16x16 output tile; bias pre-loaded into accumulator.
// q,k stored [n][h][l][d]; v stored TRANSPOSED [n][h][d][m] so the
// attention kernel's V fragment load is contiguous per lane.
__global__ void k_qkv(const _Float16* __restrict__ xn, const _Float16* __restrict__ wq,
                      const float* __restrict__ bqkv, _Float16* __restrict__ qb,
                      _Float16* __restrict__ kb, _Float16* __restrict__ vbT) {
  int wave = threadIdx.x >> 5;
  int lane = threadIdx.x & 31;
  int bid = blockIdx.x;
  int ng = bid % 6;
  int mt = (bid / 6) & 15;
  int n  = bid / 96;
  int nt = ng * 8 + wave;                 // 0..47
  const _Float16* A = xn + (size_t)n * L_TOK * CH;
  int col = nt * 16 + (lane & 15);        // 0..767
  float bias = bqkv[col];
  v8f acc;
#pragma unroll
  for (int r = 0; r < 8; ++r) acc[r] = bias;
#pragma unroll
  for (int kk = 0; kk < 8; ++kk) {
    v16h a   = load_a(A, CH, mt * 16, kk * 32);
    v16h bfr = load_b_nk(wq, CH, nt * 16, kk * 32);
    acc = wmma_f32(a, bfr, acc);
  }
  int s = col >> 8, h = (col >> 5) & 7, d = col & 31;   // uniform s,h per wave-tile
  int hi8 = ((lane >> 4) & 1) * 8;
  if (s == 2) {
    // transposed store: vT[n][h][d][m], 8 consecutive m per lane
    _Float16* dstT = vbT + (size_t)(n * NH + h) * HD * L_TOK
                   + (size_t)d * L_TOK + mt * 16 + hi8;
#pragma unroll
    for (int r = 0; r < 8; ++r) dstT[r] = (_Float16)acc[r];
  } else {
    _Float16* dst = ((s == 0) ? qb : kb)
                  + (size_t)(n * NH + h) * L_TOK * HD + d;
#pragma unroll
    for (int r = 0; r < 8; ++r) {
      int lrow = mt * 16 + r + hi8;
      dst[(size_t)lrow * HD] = (_Float16)acc[r];
    }
  }
}

// ---------------- K3: attention per (window, head, 16-row q tile) ----------
// scores (K=32 => 1 wmma per tile) -> softmax f32 in LDS -> attn@V (K=256).
__global__ void k_attn(const _Float16* __restrict__ qb, const _Float16* __restrict__ kb,
                       const _Float16* __restrict__ vbT, _Float16* __restrict__ ao) {
  __shared__ float    sc[16][256];
  __shared__ _Float16 at[16][256];
  int lane = threadIdx.x & 31;
  int lt = blockIdx.x & 15;
  int h  = (blockIdx.x >> 4) & 7;
  int n  = blockIdx.x >> 7;
  const _Float16* qbase = qb  + (size_t)(n * NH + h) * L_TOK * HD;
  const _Float16* kbase = kb  + (size_t)(n * NH + h) * L_TOK * HD;
  const _Float16* vbase = vbT + (size_t)(n * NH + h) * HD * L_TOK; // [d][m]
  int hi8 = ((lane >> 4) & 1) * 8;

  // scores = q(16x32) x k^T(32x256); column m of B = row m of k
  v16h aq = load_a(qbase, HD, lt * 16, 0);
#pragma unroll
  for (int mt2 = 0; mt2 < 16; ++mt2) {
    v16h bk = load_b_nk(kbase, HD, mt2 * 16, 0);
    v8f z = {};
    v8f scv = wmma_f32(aq, bk, z);
#pragma unroll
    for (int r = 0; r < 8; ++r)
      sc[r + hi8][mt2 * 16 + (lane & 15)] = scv[r] * SCALE_F;
  }
  __syncthreads();

  // row softmax (lanes 0..15, one row each); reconverges before next wmma
  if (lane < 16) {
    float mx = -1e30f;
    for (int m = 0; m < 256; ++m) mx = fmaxf(mx, sc[lane][m]);
    float ssum = 0.f;
    for (int m = 0; m < 256; ++m) {
      float e = __expf(sc[lane][m] - mx);
      sc[lane][m] = e; ssum += e;
    }
    float inv = 1.f / ssum;
    for (int m = 0; m < 256; ++m) at[lane][m] = (_Float16)(sc[lane][m] * inv);
  }
  __syncthreads();

  // out(16x32) = attn(16x256) x v(256x32); vT is [N=d][K=m] -> load_b_nk
#pragma unroll
  for (int dt = 0; dt < 2; ++dt) {
    v8f acc = {};
#pragma unroll
    for (int s = 0; s < 8; ++s) {
      v16h a  = load_a(&at[0][0], 256, 0, s * 32);
      v16h bv = load_b_nk(vbase, L_TOK, dt * 16, s * 32);
      acc = wmma_f32(a, bv, acc);
    }
    _Float16* dst = ao + (size_t)(n * L_TOK) * CH + h * HD + dt * 16 + (lane & 15);
#pragma unroll
    for (int r = 0; r < 8; ++r) {
      int lrow = lt * 16 + r + hi8;
      dst[(size_t)lrow * CH] = (_Float16)acc[r];
    }
  }
}

// ---------------- K4: proj GEMM + bias + residual + window scatter ---------
__global__ void k_proj(const _Float16* __restrict__ ao, const _Float16* __restrict__ wp,
                       const float* __restrict__ bp, const float* __restrict__ xw,
                       float* __restrict__ out) {
  int wave = threadIdx.x >> 5;
  int lane = threadIdx.x & 31;
  int ng = blockIdx.x & 1;
  int mt = (blockIdx.x >> 1) & 15;
  int n  = blockIdx.x >> 5;
  int nt = ng * 8 + wave;                 // 0..15
  const _Float16* A = ao + (size_t)n * L_TOK * CH;
  int col = nt * 16 + (lane & 15);        // output channel c
  float bias = bp[col];
  v8f acc;
#pragma unroll
  for (int r = 0; r < 8; ++r) acc[r] = bias;
#pragma unroll
  for (int kk = 0; kk < 8; ++kk) {
    v16h a   = load_a(A, CH, mt * 16, kk * 32);
    v16h bfr = load_b_nk(wp, CH, nt * 16, kk * 32);
    acc = wmma_f32(a, bfr, acc);
  }
  int b = n >> 6, nh = (n >> 3) & 7, nw = n & 7;
  int hi8 = ((lane >> 4) & 1) * 8;
#pragma unroll
  for (int r = 0; r < 8; ++r) {
    int l = mt * 16 + r + hi8;
    int t = l >> 6, wy = (l >> 3) & 7, wx = l & 7;
    float val = acc[r] + xw[((size_t)n * L_TOK + l) * CH + col];
    size_t oidx = ((size_t)(b * T_ + t) * CH + col) * (IMG_HW * IMG_HW)
                + (size_t)(nh * 8 + wy) * IMG_HW + (nw * 8 + wx);
    out[oidx] = val;
  }
}

// ---------------- host launch ----------------
extern "C" void kernel_launch(void* const* d_in, const int* in_sizes, int n_in,
                              void* d_out, int out_size, void* d_ws, size_t ws_size,
                              hipStream_t stream) {
  const float* x      = (const float*)d_in[0];
  const float* w_qkv  = (const float*)d_in[1];
  const float* b_qkv  = (const float*)d_in[2];
  const float* w_proj = (const float*)d_in[3];
  const float* b_proj = (const float*)d_in[4];
  const float* gamma  = (const float*)d_in[5];
  const float* beta   = (const float*)d_in[6];
  float* out = (float*)d_out;

  char* ws = (char*)d_ws;
  // workspace layout (bytes)
  float*    xw   = (float*)(ws + 0);                    // 33554432
  _Float16* xn   = (_Float16*)(ws + 33554432);          // 16777216
  _Float16* qb   = (_Float16*)(ws + 50331648);          // 16777216
  _Float16* kb   = (_Float16*)(ws + 67108864);          // 16777216
  _Float16* vbT  = (_Float16*)(ws + 83886080);          // 16777216 (transposed)
  _Float16* ao   = (_Float16*)(ws + 100663296);         // 16777216
  _Float16* wq16 = (_Float16*)(ws + 117440512);         // 393216
  _Float16* wp16 = (_Float16*)(ws + 117833728);         // 131072
  // total: 117964800 bytes

  k_cvt_w<<<1024, 256, 0, stream>>>(w_qkv, w_proj, wq16, wp16);
  k_gather_ln<<<WINDOWS * L_TOK / 8, 256, 0, stream>>>(x, gamma, beta, xw, xn);
  k_qkv<<<WINDOWS * 16 * 6, 256, 0, stream>>>(xn, wq16, b_qkv, qb, kb, vbT);
  k_attn<<<WINDOWS * NH * 16, 32, 0, stream>>>(qb, kb, vbT, ao);
  k_proj<<<WINDOWS * 16 * 2, 256, 0, stream>>>(ao, wp16, b_proj, xw, out);
}
